// HSTUBlock_26706106647266
// MI455X (gfx1250) — matrix operand order repaired and checked by reading, not compiled, not run on hardware
//
#include <hip/hip_runtime.h>
#include <hip/hip_bf16.h>
#include <math.h>

typedef __attribute__((ext_vector_type(16))) __bf16      v16bf;
typedef __attribute__((ext_vector_type(8)))  float       v8f;
typedef __attribute__((ext_vector_type(4)))  unsigned int u32x4;

constexpr int CB = 2, CS = 2048, CD = 1024, CH = 8, CDH = 128;
constexpr int CTOTAL = 4 * CD;          // 4096
constexpr int CROWS  = CB * CS;         // 4096
constexpr float CEPS = 1e-5f;

union Frag16 { v16bf v; u32x4 q[2]; };

// ---- WMMA helpers -----------------------------------------------------------

__device__ inline v8f wmma_bf16(v16bf a, v16bf b, v8f c) {
  // emits v_wmma_f32_16x16x32_bf16
  return __builtin_amdgcn_wmma_f32_16x16x32_bf16(
      /*neg_a=*/false, a, /*neg_b=*/false, b,
      /*c_mod=*/(short)0, c, /*reuse_a=*/false, /*reuse_b=*/false);
}

// A-matrix fragment: 16 rows x 32 K, row-major source (ld = row stride).
// Lane L<16: row=L, K in {k0..k0+7, k0+16..k0+23}; lanes 16-31: row=L-16, K shifted +8.
__device__ inline v16bf load_a_frag(const __bf16* __restrict__ base, int row0, int ld,
                                    int k0, int lane) {
  int r  = lane & 15;
  int kk = k0 + ((lane >> 4) << 3);
  const __bf16* p = base + (size_t)(row0 + r) * (size_t)ld + kk;
  Frag16 f;
  f.q[0] = *(const u32x4*)(p);
  f.q[1] = *(const u32x4*)(p + 16);
  return f.v;
}

// B-matrix fragment: 32 K x 16 cols. Source stored N-major: Bt[n][k], ld = K stride.
// Lane L<16: col=L, K in {k0..k0+7, k0+16..k0+23}; lanes 16-31: col=L-16, K shifted +8.
__device__ inline v16bf load_b_frag(const __bf16* __restrict__ base, int n0, int ld,
                                    int k0, int lane) {
  int nn = n0 + (lane & 15);
  int kk = k0 + ((lane >> 4) << 3);
  const __bf16* p = base + (size_t)nn * (size_t)ld + kk;
  Frag16 f;
  f.q[0] = *(const u32x4*)(p);
  f.q[1] = *(const u32x4*)(p + 16);
  return f.v;
}

__device__ inline float silu_f(float x) { return x / (1.0f + __expf(-x)); }

// ---- Kernel 1: LayerNorm(x) -> bf16 ----------------------------------------

__global__ __launch_bounds__(256) void
ln1_cast_kernel(const float* __restrict__ x, const float* __restrict__ g,
                const float* __restrict__ bb, __bf16* __restrict__ out) {
  int row = blockIdx.x;
  const float* xr = x + (size_t)row * CD;
  __shared__ float red[256];
  int t = threadIdx.x;
  float s = 0.f;
  for (int j = t; j < CD; j += 256) s += xr[j];
  red[t] = s; __syncthreads();
  for (int o = 128; o > 0; o >>= 1) { if (t < o) red[t] += red[t + o]; __syncthreads(); }
  float mean = red[0] * (1.0f / CD);
  __syncthreads();
  float vs = 0.f;
  for (int j = t; j < CD; j += 256) { float d = xr[j] - mean; vs += d * d; }
  red[t] = vs; __syncthreads();
  for (int o = 128; o > 0; o >>= 1) { if (t < o) red[t] += red[t + o]; __syncthreads(); }
  float rstd = rsqrtf(red[0] * (1.0f / CD) + CEPS);
  __bf16* orow = out + (size_t)row * CD;
  for (int j = t; j < CD; j += 256)
    orow[j] = (__bf16)((xr[j] - mean) * rstd * g[j] + bb[j]);
}

// ---- Kernel 2: transpose + cast weights fp32[K,N] -> bf16[N,K] --------------

__global__ __launch_bounds__(256) void
cvt_transpose_kernel(const float* __restrict__ w, __bf16* __restrict__ wt, int K, int N) {
  int idx = blockIdx.x * blockDim.x + threadIdx.x;
  if (idx >= K * N) return;
  int k = idx / N, n = idx % N;
  wt[(size_t)n * K + k] = (__bf16)w[(size_t)k * N + n];
}

// ---- Dense GEMM: 32x64 tile per wave, 4 waves/block -> 128x64 per block -----
// C[m,n] = epilogue(sum_k A[m,k] * Bt[n,k])
// EPI: 0 = bias + silu -> C (f32)
//      1 = bias + residual(x) -> C (f32)
// __launch_bounds__(128, 1): ~140 live VGPRs per wave (64 acc + 48 frag + addr);
// allow the full budget so nothing spills to scratch.

template <int KDIM, int EPI>
__global__ __launch_bounds__(128, 1) void
gemm_tile_kernel(const __bf16* __restrict__ A, const __bf16* __restrict__ Bt,
                 const float* __restrict__ bias, const float* __restrict__ resid,
                 float* __restrict__ C, int ldc) {
  int lane = threadIdx.x & 31, w = threadIdx.x >> 5;
  int mW = blockIdx.x * 128 + w * 32;   // 32 rows per wave
  int n0 = blockIdx.y * 64;             // 64 cols per wave

  v8f acc[2][4] = {};
  for (int k0 = 0; k0 < KDIM; k0 += 32) {
    // prefetch the weight stream a few chunks ahead (global_prefetch_b8)
    if (k0 + 128 < KDIM) {
      const __bf16* pf = Bt + (size_t)(n0 + (lane & 15)) * KDIM + (k0 + 128);
      __builtin_prefetch(pf, 0, 1);
    }
    v16bf a0 = load_a_frag(A, mW,      KDIM, k0, lane);
    v16bf a1 = load_a_frag(A, mW + 16, KDIM, k0, lane);
    v16bf b0 = load_b_frag(Bt, n0,      KDIM, k0, lane);
    v16bf b1 = load_b_frag(Bt, n0 + 16, KDIM, k0, lane);
    v16bf b2 = load_b_frag(Bt, n0 + 32, KDIM, k0, lane);
    v16bf b3 = load_b_frag(Bt, n0 + 48, KDIM, k0, lane);
    acc[0][0] = wmma_bf16(a0, b0, acc[0][0]);
    acc[0][1] = wmma_bf16(a0, b1, acc[0][1]);
    acc[0][2] = wmma_bf16(a0, b2, acc[0][2]);
    acc[0][3] = wmma_bf16(a0, b3, acc[0][3]);
    acc[1][0] = wmma_bf16(a1, b0, acc[1][0]);
    acc[1][1] = wmma_bf16(a1, b1, acc[1][1]);
    acc[1][2] = wmma_bf16(a1, b2, acc[1][2]);
    acc[1][3] = wmma_bf16(a1, b3, acc[1][3]);
  }

  int nl = lane & 15;
  int mo = (lane >> 4) << 3;
#pragma unroll
  for (int mi = 0; mi < 2; ++mi) {
#pragma unroll
    for (int ni = 0; ni < 4; ++ni) {
      int n = n0 + ni * 16 + nl;
      float bi = bias[n];
#pragma unroll
      for (int r = 0; r < 8; ++r) {
        size_t idx = (size_t)(mW + mi * 16 + mo + r) * ldc + n;
        float v = acc[mi][ni][r] + bi;
        if (EPI == 0) {
          C[idx] = silu_f(v);
        } else {
          C[idx] = v + resid[idx];
        }
      }
    }
  }
}

// ---- Kernel 4: RoPE q,k -> bf16 [B,H,S,DH]; v -> bf16 transposed [B,H,DH,S] -

__global__ __launch_bounds__(256) void
rope_pack_kernel(const float* __restrict__ uvqk,
                 __bf16* __restrict__ qb, __bf16* __restrict__ kb,
                 __bf16* __restrict__ vt) {
  constexpr int HALF = CDH / 2;
  int idx = blockIdx.x * blockDim.x + threadIdx.x;
  if (idx >= CB * CH * CS * HALF) return;
  int i = idx % HALF;
  int s = (idx / HALF) % CS;
  int h = (idx / (HALF * CS)) % CH;
  int b = idx / (HALF * CS * CH);
  const float* ur = uvqk + ((size_t)b * CS + s) * CTOTAL;
  float inv = powf(10000.0f, -(float)(2 * i) / (float)CDH);
  float sn, cs;
  __sincosf((float)s * inv, &sn, &cs);
  int bh = b * CH + h;
  size_t qoff = ((size_t)bh * CS + s) * CDH;
  {
    float xe = ur[2 * CD + h * CDH + 2 * i], xo = ur[2 * CD + h * CDH + 2 * i + 1];
    qb[qoff + 2 * i]     = (__bf16)(xe * cs - xo * sn);
    qb[qoff + 2 * i + 1] = (__bf16)(xe * sn + xo * cs);
  }
  {
    float xe = ur[3 * CD + h * CDH + 2 * i], xo = ur[3 * CD + h * CDH + 2 * i + 1];
    kb[qoff + 2 * i]     = (__bf16)(xe * cs - xo * sn);
    kb[qoff + 2 * i + 1] = (__bf16)(xe * sn + xo * cs);
  }
  {
    size_t vb = (size_t)bh * CDH * CS;
    vt[vb + (size_t)(2 * i)     * CS + s] = (__bf16)ur[CD + h * CDH + 2 * i];
    vt[vb + (size_t)(2 * i + 1) * CS + s] = (__bf16)ur[CD + h * CDH + 2 * i + 1];
  }
}

// ---- Kernel 5: fused SiLU attention -----------------------------------------
// grid = (S/64, B*H), block = 128 (4 waves, one 16-row q-tile each).
// __launch_bounds__(128, 1): peak ~150 live VGPRs (32 q-frag + 64 acc + score
// temps); allow full budget so base pointers / fragments are never spilled.

__global__ __launch_bounds__(128, 1) void
attn_kernel(const __bf16* __restrict__ qb, const __bf16* __restrict__ kb,
            const __bf16* __restrict__ vt, const unsigned char* __restrict__ mask,
            float* __restrict__ ao) {
  __shared__ float sc[4][16][32];
  int lane = threadIdx.x & 31, w = threadIdx.x >> 5;
  int bh = blockIdx.y;
  int b  = bh / CH, h = bh % CH;
  const __bf16* qh = qb + (size_t)bh * CS * CDH;
  const __bf16* kh = kb + (size_t)bh * CS * CDH;
  const __bf16* vh = vt + (size_t)bh * CDH * CS;
  const unsigned char* mrow = mask + (size_t)b * CS;
  int q0 = blockIdx.x * 64 + w * 16;

  v16bf qf[4];
#pragma unroll
  for (int g = 0; g < 4; ++g) qf[g] = load_a_frag(qh, q0, CDH, g * 32, lane);

  v8f acc[8] = {};
  const float rscale = 0.0883883476483184f;  // 1/sqrt(128)
  int n  = lane & 15;
  int mo = (lane >> 4) << 3;
  // uniform across all 4 waves so __syncthreads is safe; extra chunks get fully
  // masked to -30 -> silu(-30) ~ 0, matching the reference's masking semantics.
  int nChunks = 2 * blockIdx.x + 2;

  for (int c = 0; c < nChunks; ++c) {
    int kBase = c * 32;
    v8f s0 = {}, s1 = {};
#pragma unroll
    for (int g = 0; g < 4; ++g) {
      v16bf kf0 = load_b_frag(kh, kBase,      CDH, g * 32, lane);
      v16bf kf1 = load_b_frag(kh, kBase + 16, CDH, g * 32, lane);
      s0 = wmma_bf16(qf[g], kf0, s0);
      s1 = wmma_bf16(qf[g], kf1, s1);
    }
    // scale / mask / clip / silu, stage through LDS to re-swizzle C->A layout
#pragma unroll
    for (int r = 0; r < 8; ++r) {
      int qi = q0 + mo + r;
      bool qok = mrow[qi] != 0;
      {
        int kj = kBase + n;
        float v = s0[r] * rscale;
        bool ok = qok && (kj <= qi) && (mrow[kj] != 0);
        v = ok ? v : -30.0f;
        v = fminf(fmaxf(v, -30.0f), 30.0f);
        sc[w][mo + r][n] = silu_f(v);
      }
      {
        int kj = kBase + 16 + n;
        float v = s1[r] * rscale;
        bool ok = qok && (kj <= qi) && (mrow[kj] != 0);
        v = ok ? v : -30.0f;
        v = fminf(fmaxf(v, -30.0f), 30.0f);
        sc[w][mo + r][16 + n] = silu_f(v);
      }
    }
    __syncthreads();
    v16bf af;
    {
      int r  = lane & 15;
      int kk = (lane >> 4) << 3;
#pragma unroll
      for (int e = 0; e < 8; ++e) af[e]     = (__bf16)sc[w][r][kk + e];
#pragma unroll
      for (int e = 0; e < 8; ++e) af[8 + e] = (__bf16)sc[w][r][kk + 16 + e];
    }
#pragma unroll
    for (int g = 0; g < 8; ++g) {
      v16bf vf = load_b_frag(vh, g * 16, CS, kBase, lane);  // vT[dh][s]
      acc[g] = wmma_bf16(af, vf, acc[g]);
    }
    __syncthreads();
  }

#pragma unroll
  for (int g = 0; g < 8; ++g) {
#pragma unroll
    for (int r = 0; r < 8; ++r) {
      int m = q0 + mo + r;
      ao[((size_t)b * CS + m) * CD + h * CDH + g * 16 + n] = acc[g][r];
    }
  }
}

// ---- Kernel 6: LN2 + build concat [u, ao, ln2(ao)*u] as bf16 ----------------

__global__ __launch_bounds__(256) void
ln2_cat_kernel(const float* __restrict__ uvqk, const float* __restrict__ ao,
               const float* __restrict__ g, const float* __restrict__ bb,
               __bf16* __restrict__ cat) {
  int row = blockIdx.x;
  const float* ar = ao + (size_t)row * CD;
  const float* ur = uvqk + (size_t)row * CTOTAL;  // u = cols [0, D)
  __shared__ float red[256];
  int t = threadIdx.x;
  float s = 0.f;
  for (int j = t; j < CD; j += 256) s += ar[j];
  red[t] = s; __syncthreads();
  for (int o = 128; o > 0; o >>= 1) { if (t < o) red[t] += red[t + o]; __syncthreads(); }
  float mean = red[0] * (1.0f / CD);
  __syncthreads();
  float vs = 0.f;
  for (int j = t; j < CD; j += 256) { float d = ar[j] - mean; vs += d * d; }
  red[t] = vs; __syncthreads();
  for (int o = 128; o > 0; o >>= 1) { if (t < o) red[t] += red[t + o]; __syncthreads(); }
  float rstd = rsqrtf(red[0] * (1.0f / CD) + CEPS);
  __bf16* cr = cat + (size_t)row * (3 * CD);
  for (int j = t; j < CD; j += 256) {
    float u  = ur[j];
    float a  = ar[j];
    float na = (a - mean) * rstd * g[j] + bb[j];
    cr[j]          = (__bf16)u;
    cr[CD + j]     = (__bf16)a;
    cr[2 * CD + j] = (__bf16)(na * u);
  }
}

// ---- Host launcher ----------------------------------------------------------

extern "C" void kernel_launch(void* const* d_in, const int* in_sizes, int n_in,
                              void* d_out, int out_size, void* d_ws, size_t ws_size,
                              hipStream_t stream) {
  const float* x            = (const float*)d_in[0];
  const unsigned char* mask = (const unsigned char*)d_in[1];  // jnp bool = 1 byte
  const float* ln1_g  = (const float*)d_in[2];
  const float* ln1_b  = (const float*)d_in[3];
  const float* w_uvqk = (const float*)d_in[4];
  const float* b_uvqk = (const float*)d_in[5];
  const float* ln2_g  = (const float*)d_in[6];
  const float* ln2_b  = (const float*)d_in[7];
  const float* w_out  = (const float*)d_in[8];
  const float* b_out  = (const float*)d_in[9];
  float* out = (float*)d_out;

  char* ws = (char*)d_ws;
  const size_t MB = 1ull << 20;
  __bf16* normed  = (__bf16*)(ws + 0 * MB);    // 8 MiB  [4096,1024] bf16
  __bf16* wT_uvqk = (__bf16*)(ws + 8 * MB);    // 8 MiB  [4096,1024] bf16 (N-major)
  __bf16* wT_out  = (__bf16*)(ws + 16 * MB);   // 6 MiB  [1024,3072] bf16 (N-major)
  float*  uvqk    = (float*)(ws + 22 * MB);    // 64 MiB [4096,4096] f32
  __bf16* qb      = (__bf16*)(ws + 86 * MB);   // 8 MiB  [B,H,S,DH] bf16
  __bf16* kb      = (__bf16*)(ws + 94 * MB);   // 8 MiB
  __bf16* vtb     = (__bf16*)(ws + 102 * MB);  // 8 MiB  [B,H,DH,S] bf16
  float*  ao      = (float*)(ws + 110 * MB);   // 16 MiB [4096,1024] f32
  __bf16* cat     = (__bf16*)(ws + 126 * MB);  // 24 MiB [4096,3072] bf16

  ln1_cast_kernel<<<CROWS, 256, 0, stream>>>(x, ln1_g, ln1_b, normed);
  cvt_transpose_kernel<<<(CD * CTOTAL + 255) / 256, 256, 0, stream>>>(w_uvqk, wT_uvqk, CD, CTOTAL);
  cvt_transpose_kernel<<<(3 * CD * CD + 255) / 256, 256, 0, stream>>>(w_out, wT_out, 3 * CD, CD);

  // GEMM1: [4096,1024] x [1024,4096] -> silu -> uvqk
  gemm_tile_kernel<CD, 0><<<dim3(CROWS / 128, CTOTAL / 64), 128, 0, stream>>>(
      normed, wT_uvqk, b_uvqk, nullptr, uvqk, CTOTAL);

  rope_pack_kernel<<<(CB * CH * CS * (CDH / 2) + 255) / 256, 256, 0, stream>>>(uvqk, qb, kb, vtb);
  attn_kernel<<<dim3(CS / 64, CB * CH), 128, 0, stream>>>(qb, kb, vtb, mask, ao);
  ln2_cat_kernel<<<CROWS, 256, 0, stream>>>(uvqk, ao, ln2_g, ln2_b, cat);

  // GEMM2: [4096,3072] x [3072,1024] + bias + residual -> out
  gemm_tile_kernel<3 * CD, 1><<<dim3(CROWS / 128, CD / 64), 128, 0, stream>>>(
      cat, wT_out, b_out, x, out, CD);
}
